// DGCRN_7593502179665
// MI455X (gfx1250) — matrix operand order repaired and checked by reading, not compiled, not run on hardware
//
#include <hip/hip_runtime.h>
#include <math.h>

// ---------------------------------------------------------------------------
// Problem constants (fixed by the reference: B=64, T=12, N=2048, Fi=1, H=64)
// ---------------------------------------------------------------------------
#define B_   64
#define T_   12
#define N_   2048
#define H_   64
#define F_   65                 // Fi + H
#define NC_  (B_ * F_)          // 4160  columns of the hop GEMM output
#define IN_  195                // (GCN_DEPTH+1)*(Fi+H)
#define KP_  224                // IN_ padded up to a multiple of 32
#define M2_  (B_ * N_)          // 131072 rows of the gate GEMMs
#define HOR_ 12

typedef __attribute__((ext_vector_type(16))) __bf16 v16bf;
typedef __attribute__((ext_vector_type(8)))  float  v8f;

union FragB16 { v16bf v; unsigned int u[8]; };

static __device__ __forceinline__ unsigned short f2bf(float f) {
  unsigned int u = __float_as_uint(f);
  u += 0x7FFFu + ((u >> 16) & 1u);          // round-to-nearest-even
  return (unsigned short)(u >> 16);
}
static __device__ __forceinline__ float bf2f(unsigned short s) {
  return __uint_as_float(((unsigned int)s) << 16);
}

// ---------------------------------------------------------------------------
// Setup kernels (run once per launch)
// ---------------------------------------------------------------------------
__global__ void k_rownorm(const float* __restrict__ adj,
                          unsigned short* __restrict__ out) {
  int row = blockIdx.x, tid = threadIdx.x;
  __shared__ float red[256];
  const float* r = adj + (size_t)row * N_;
  float s = 0.f;
  for (int i = tid; i < N_; i += 256) s += r[i];
  red[tid] = s; __syncthreads();
  for (int st = 128; st; st >>= 1) {
    if (tid < st) red[tid] += red[tid + st];
    __syncthreads();
  }
  float inv = 1.f / (red[0] + 1e-8f);
  for (int i = tid; i < N_; i += 256)
    out[(size_t)row * N_ + i] = f2bf(r[i] * inv);
}

__global__ void k_logits(const float* __restrict__ e1,
                         const float* __restrict__ e2,
                         float* __restrict__ lg) {
  int idx = blockIdx.x * 256 + threadIdx.x;        // exactly N_*N_ threads
  int n = idx >> 11, m = idx & (N_ - 1);
  float s = 0.f;
#pragma unroll
  for (int d = 0; d < 16; ++d) s += e1[n * 16 + d] * e2[d * N_ + m];
  lg[idx] = s > 0.f ? s : 0.f;                     // relu
}

__global__ void k_softmax_row(const float* __restrict__ lg,
                              unsigned short* __restrict__ out) {
  int row = blockIdx.x, tid = threadIdx.x;
  __shared__ float red[256];
  const float* lr = lg + (size_t)row * N_;
  float mx = -1e30f;
  for (int i = tid; i < N_; i += 256) mx = fmaxf(mx, lr[i]);
  red[tid] = mx; __syncthreads();
  for (int st = 128; st; st >>= 1) {
    if (tid < st) red[tid] = fmaxf(red[tid], red[tid + st]);
    __syncthreads();
  }
  mx = red[0]; __syncthreads();
  float sm = 0.f;
  for (int i = tid; i < N_; i += 256) sm += __expf(lr[i] - mx);
  red[tid] = sm; __syncthreads();
  for (int st = 128; st; st >>= 1) {
    if (tid < st) red[tid] += red[tid + st];
    __syncthreads();
  }
  float inv = 1.f / red[0];
  for (int i = tid; i < N_; i += 256)
    out[(size_t)row * N_ + i] = f2bf(__expf(lr[i] - mx) * inv);
}

// W [IN_ x H_] -> Wt (bf16, transposed+padded) [H_ x KP_]
__global__ void k_pad_w(const float* __restrict__ src,
                        unsigned short* __restrict__ dst) {
  int idx = blockIdx.x * 256 + threadIdx.x;
  if (idx >= H_ * KP_) return;
  int h = idx / KP_, k = idx % KP_;
  dst[idx] = (k < IN_) ? f2bf(src[k * H_ + h]) : (unsigned short)0;
}

// ---------------------------------------------------------------------------
// Per-step elementwise / packing kernels
// ---------------------------------------------------------------------------
__global__ void k_reduce_sum(const float* __restrict__ xsrc, int t, int T,
                             float* __restrict__ sum) {
  __shared__ float red[256];
  int tid = threadIdx.x;
  float s = 0.f;
  for (int i = blockIdx.x * 256 + tid; i < B_ * N_; i += gridDim.x * 256) {
    int b = i / N_, n = i % N_;
    s += xsrc[((size_t)b * T + t) * N_ + n];
  }
  red[tid] = s; __syncthreads();
  for (int st = 128; st; st >>= 1) {
    if (tid < st) red[tid] += red[tid + st];
    __syncthreads();
  }
  if (tid == 0) atomicAdd(sum, red[0]);
}

__global__ void k_blend_adj(const unsigned short* __restrict__ sn,
                            const unsigned short* __restrict__ ad,
                            unsigned short* __restrict__ out,
                            const float* __restrict__ sum,
                            const float* __restrict__ cs,
                            const float* __restrict__ cb) {
  int idx = blockIdx.x * 256 + threadIdx.x;        // exactly N_*N_ threads
  float mean = sum[0] * (1.f / (float)(B_ * N_));
  float g = 1.f / (1.f + __expf(-(cs[0] * mean + cb[0])));
  float v = g * bf2f(sn[idx]) + (1.f - g) * bf2f(ad[idx]);
  out[idx] = f2bf(v);
}

// XhT[c][m] with c = b*F_+f : (f==0 ? x_t[b,m] : (r?r*h:h)[b,m,f-1]) as bf16
__global__ void k_pack_xh(const float* __restrict__ xsrc, int t, int T,
                          const float* __restrict__ h,
                          const float* __restrict__ r,
                          unsigned short* __restrict__ out) {
  int idx = blockIdx.x * 256 + threadIdx.x;
  if (idx >= NC_ * N_) return;
  int c = idx / N_, m = idx % N_;
  int b = c / F_, f = c % F_;
  float v;
  if (f == 0) {
    v = xsrc[((size_t)b * T + t) * N_ + m];
  } else {
    size_t hi = ((size_t)b * N_ + m) * H_ + (f - 1);
    v = h[hi];
    if (r) v *= r[hi];
  }
  out[idx] = f2bf(v);
}

// G[b*N+n, c] : 0..64 -> XhT, 65..129 -> hop1T(bf16), 130..194 -> hop2T(f32)
__global__ void k_pack_g(const unsigned short* __restrict__ xhT,
                         const unsigned short* __restrict__ h1T,
                         const float* __restrict__ h2T,
                         unsigned short* __restrict__ g) {
  int idx = blockIdx.x * 256 + threadIdx.x;        // exactly M2_*KP_ threads
  int row = idx / KP_, c = idx % KP_;
  int b = row / N_, n = row % N_;
  unsigned short u;
  if (c < F_)
    u = xhT[(size_t)(b * F_ + c) * N_ + n];
  else if (c < 2 * F_)
    u = h1T[(size_t)(b * F_ + (c - F_)) * N_ + n];
  else if (c < IN_)
    u = f2bf(h2T[(size_t)(b * F_ + (c - 2 * F_)) * N_ + n]);
  else
    u = 0;
  g[idx] = u;
}

// ---------------------------------------------------------------------------
// Tiled bf16 WMMA GEMM: C[M x Ncols] = A[M x K](bf16) * B[K x Ncols]
// B supplied TRANSPOSED: Bt[Ncols x K] row-major (our packers produce this).
// Block tile 128x64x32, 8 wave32s, each wave -> 32x32 via 2x2 WMMA 16x16x32.
// mode 0: f32 store            (C[row][col], ldc = Ncols stride)
// mode 1: sigmoid(x+bias[col]) (f32, normal layout)
// mode 2: tanh(x+bias[col])    (f32, normal layout)
// mode 3: bf16 store TRANSPOSED (Ct[col][row], ldc = M stride)
// mode 4: f32  store TRANSPOSED (Ct[col][row], ldc = M stride)
// M%128==0, Ncols%64==0, K%32==0 guaranteed by caller -> EXEC all-ones.
// ---------------------------------------------------------------------------
#define BM 128
#define BN 64
#define BK 32
#define LDS_S 40                // padded LDS stride (elements)

__global__ __launch_bounds__(256)
void k_gemm_bf16(const unsigned short* __restrict__ A, int lda,
                 const unsigned short* __restrict__ Bt, int ldbt,
                 void* __restrict__ Cout, int ldc, int K,
                 const float* __restrict__ bias, int mode) {
  __shared__ __align__(16) unsigned short As[BM * LDS_S];
  __shared__ __align__(16) unsigned short Bs[BN * LDS_S];

  int t = threadIdx.x;
  int lane = t & 31, wave = t >> 5;
  int wr = wave & 3, wc = wave >> 2;
  int rowBlock = blockIdx.x * BM;
  int colBlock = blockIdx.y * BN;

  v8f acc[2][2];
#pragma unroll
  for (int i = 0; i < 2; ++i)
#pragma unroll
    for (int j = 0; j < 2; ++j)
      acc[i][j] = (v8f){0.f, 0.f, 0.f, 0.f, 0.f, 0.f, 0.f, 0.f};

  // cooperative tile-load coordinates
  int arow = t >> 1, acol = (t & 1) * 16;  // A tile: 16 bf16 per thread
  int bcol = t >> 2, bkof = (t & 3) * 8;   // Bt tile: 8 bf16 per thread

  for (int k0 = 0; k0 < K; k0 += BK) {
    __syncthreads();
    const unsigned short* ga =
        A + (size_t)(rowBlock + arow) * lda + k0 + acol;
    *(uint4*)(&As[arow * LDS_S + acol])     = *(const uint4*)ga;
    *(uint4*)(&As[arow * LDS_S + acol + 8]) = *(const uint4*)(ga + 8);
    const unsigned short* gb =
        Bt + (size_t)(colBlock + bcol) * ldbt + k0 + bkof;
    *(uint4*)(&Bs[bcol * LDS_S + bkof]) = *(const uint4*)gb;
    if (k0 + BK < K) {                       // prefetch next K tile
      __builtin_prefetch(ga + BK, 0, 1);
      __builtin_prefetch(gb + BK, 0, 1);
    }
    __syncthreads();

    int half = lane >> 4;
    FragB16 af[2], bfr[2];
#pragma unroll
    for (int i = 0; i < 2; ++i) {
      const unsigned short* base =
          &As[(wr * 32 + i * 16 + (lane & 15)) * LDS_S];
#pragma unroll
      for (int j = 0; j < 8; ++j) {
        int kk = 2 * j + 16 * half;                       // contiguous k, matches B
        af[i].u[j] = *(const unsigned int*)(base + kk);
      }
    }
#pragma unroll
    for (int j2 = 0; j2 < 2; ++j2) {
      const unsigned short* base =
          &Bs[(wc * 32 + j2 * 16 + (lane & 15)) * LDS_S];
#pragma unroll
      for (int j = 0; j < 8; ++j) {
        int kk = 2 * j + 16 * half;                       // contiguous k, matches A
        bfr[j2].u[j] = *(const unsigned int*)(base + kk); // contiguous pairs
      }
    }
#pragma unroll
    for (int i = 0; i < 2; ++i)
#pragma unroll
      for (int j2 = 0; j2 < 2; ++j2)
        acc[i][j2] = __builtin_amdgcn_wmma_f32_16x16x32_bf16(
            false, af[i].v, false, bfr[j2].v, (short)0, acc[i][j2],
            false, false);
  }

  // epilogue: C/D layout = row (v + 8*half), col (lane&15)
  int half = lane >> 4;
  int cn = lane & 15;
#pragma unroll
  for (int i = 0; i < 2; ++i) {
#pragma unroll
    for (int j2 = 0; j2 < 2; ++j2) {
      int row0 = rowBlock + wr * 32 + i * 16 + 8 * half;
      int col = colBlock + wc * 32 + j2 * 16 + cn;
      if (mode == 3) {
        unsigned short* Ct = (unsigned short*)Cout;
#pragma unroll
        for (int v = 0; v < 8; ++v)
          Ct[(size_t)col * ldc + row0 + v] = f2bf(acc[i][j2][v]);
      } else if (mode == 4) {
        float* Ct = (float*)Cout;
#pragma unroll
        for (int v = 0; v < 8; ++v)
          Ct[(size_t)col * ldc + row0 + v] = acc[i][j2][v];
      } else {
        float* Cf = (float*)Cout;
#pragma unroll
        for (int v = 0; v < 8; ++v) {
          float x = acc[i][j2][v];
          if (mode == 1)      x = 1.f / (1.f + __expf(-(x + bias[col])));
          else if (mode == 2) x = tanhf(x + bias[col]);
          Cf[(size_t)(row0 + v) * ldc + col] = x;
        }
      }
    }
  }
}

// ---------------------------------------------------------------------------
// GRU update + layernorm. One wave32 per (b,n) row of H=64; 2 elems/lane.
// ---------------------------------------------------------------------------
__global__ void k_update_ln(const float* __restrict__ z,
                            const float* __restrict__ ht,
                            const float* __restrict__ hp,
                            const float* __restrict__ gamma,
                            const float* __restrict__ beta,
                            float* __restrict__ hout) {
  int row = blockIdx.x * 8 + threadIdx.y;
  int lane = threadIdx.x;
  size_t b0 = (size_t)row * H_ + lane;
  size_t b1 = b0 + 32;
  float z0 = z[b0], z1 = z[b1];
  float h0 = (1.f - z0) * hp[b0] + z0 * ht[b0];
  float h1 = (1.f - z1) * hp[b1] + z1 * ht[b1];
  float s = h0 + h1, q = h0 * h0 + h1 * h1;
#pragma unroll
  for (int m = 16; m >= 1; m >>= 1) {
    s += __shfl_xor(s, m, 32);
    q += __shfl_xor(q, m, 32);
  }
  float mu = s * (1.f / H_);
  float var = q * (1.f / H_) - mu * mu;
  float rs = rsqrtf(var + 1e-5f);
  hout[b0] = (h0 - mu) * rs * gamma[lane] + beta[lane];
  hout[b1] = (h1 - mu) * rs * gamma[lane + 32] + beta[lane + 32];
}

// out[b, step, n] = h[b,n,:] . proj_w + proj_b ; also feeds decoder input y
__global__ void k_project(const float* __restrict__ h,
                          const float* __restrict__ pw,
                          const float* __restrict__ pb,
                          float* __restrict__ out,
                          float* __restrict__ y, int step) {
  int row = blockIdx.x * 8 + threadIdx.y;   // b*N + n
  int lane = threadIdx.x;
  size_t b0 = (size_t)row * H_ + lane;
  float s = h[b0] * pw[lane] + h[b0 + 32] * pw[lane + 32];
#pragma unroll
  for (int m = 16; m >= 1; m >>= 1) s += __shfl_xor(s, m, 32);
  if (lane == 0) {
    float v = s + pb[0];
    int b = row / N_, n = row % N_;
    out[((size_t)b * HOR_ + step) * N_ + n] = v;
    y[row] = v;
  }
}

// ---------------------------------------------------------------------------
// Host orchestration
// ---------------------------------------------------------------------------
extern "C" void kernel_launch(void* const* d_in, const int* in_sizes, int n_in,
                              void* d_out, int out_size, void* d_ws,
                              size_t ws_size, hipStream_t stream) {
  (void)in_sizes; (void)n_in; (void)out_size; (void)ws_size;

  const float* X          = (const float*)d_in[0];
  const float* static_adj = (const float*)d_in[1];
  const float* e1         = (const float*)d_in[2];
  const float* e2         = (const float*)d_in[3];
  const float* cs         = (const float*)d_in[4];
  const float* cb         = (const float*)d_in[5];
  const float* enc_wz = (const float*)d_in[6];  const float* enc_bz = (const float*)d_in[7];
  const float* enc_wr = (const float*)d_in[8];  const float* enc_br = (const float*)d_in[9];
  const float* enc_wh = (const float*)d_in[10]; const float* enc_bh = (const float*)d_in[11];
  const float* enc_g  = (const float*)d_in[12]; const float* enc_b  = (const float*)d_in[13];
  const float* dec_wz = (const float*)d_in[14]; const float* dec_bz = (const float*)d_in[15];
  const float* dec_wr = (const float*)d_in[16]; const float* dec_br = (const float*)d_in[17];
  const float* dec_wh = (const float*)d_in[18]; const float* dec_bh = (const float*)d_in[19];
  const float* dec_g  = (const float*)d_in[20]; const float* dec_b  = (const float*)d_in[21];
  const float* proj_w = (const float*)d_in[22]; const float* proj_b = (const float*)d_in[23];
  float* out = (float*)d_out;

  // ---- workspace carve (~320 MB total) ----
  char* p = (char*)d_ws;
  auto carve = [&](size_t bytes) -> void* {
    void* r = (void*)p;
    p += (bytes + 255) & ~(size_t)255;
    return r;
  };
  unsigned short* sn_bf  = (unsigned short*)carve((size_t)N_ * N_ * 2);
  unsigned short* ad_bf  = (unsigned short*)carve((size_t)N_ * N_ * 2);
  unsigned short* adj_bf = (unsigned short*)carve((size_t)N_ * N_ * 2);
  unsigned short* wpad   = (unsigned short*)carve((size_t)6 * KP_ * H_ * 2);
  float* h_a    = (float*)carve((size_t)M2_ * H_ * 4);
  float* h_b    = (float*)carve((size_t)M2_ * H_ * 4);
  unsigned short* xh_bf  = (unsigned short*)carve((size_t)N_ * NC_ * 2);
  unsigned short* hop_bf = (unsigned short*)carve((size_t)N_ * NC_ * 2);
  float* hop2_f = (float*)carve((size_t)N_ * NC_ * 4);   // also logits scratch
  unsigned short* g_bf   = (unsigned short*)carve((size_t)M2_ * KP_ * 2);
  float* zbuf   = (float*)carve((size_t)M2_ * H_ * 4);
  float* rbuf   = (float*)carve((size_t)M2_ * H_ * 4);
  float* htbuf  = (float*)carve((size_t)M2_ * H_ * 4);
  float* ybuf   = (float*)carve((size_t)B_ * N_ * 4);
  float* gsum   = (float*)carve(256);

  unsigned short* wp[6] = {wpad,                wpad + KP_ * H_,
                           wpad + 2 * KP_ * H_, wpad + 3 * KP_ * H_,
                           wpad + 4 * KP_ * H_, wpad + 5 * KP_ * H_};

  // ---- one-time setup ----
  k_rownorm<<<N_, 256, 0, stream>>>(static_adj, sn_bf);
  k_logits<<<(N_ * N_) / 256, 256, 0, stream>>>(e1, e2, hop2_f);
  k_softmax_row<<<N_, 256, 0, stream>>>(hop2_f, ad_bf);
  const float* wsrc[6] = {enc_wz, enc_wr, enc_wh, dec_wz, dec_wr, dec_wh};
  for (int i = 0; i < 6; ++i)
    k_pad_w<<<(KP_ * H_ + 255) / 256, 256, 0, stream>>>(wsrc[i], wp[i]);

  hipMemsetAsync(h_a, 0, (size_t)M2_ * H_ * 4, stream);

  dim3 gHop(N_ / BM, NC_ / BN);     // 16 x 65
  dim3 gGate(M2_ / BM, H_ / BN);    // 1024 x 1
  dim3 rowsBlk(32, 8);
  int nPackXh = (NC_ * N_ + 255) / 256;
  int nPackG  = (M2_ * KP_) / 256;

  auto cell = [&](const float* xsrc, int t, int T,
                  const unsigned short* wz, const float* bz,
                  const unsigned short* wr, const float* br,
                  const unsigned short* wh, const float* bh,
                  const float* gamma, const float* beta,
                  const float* h_in, float* h_out) {
    // gate scalar -> blended adjacency (bf16 A matrix)
    hipMemsetAsync(gsum, 0, 4, stream);
    k_reduce_sum<<<256, 256, 0, stream>>>(xsrc, t, T, gsum);
    k_blend_adj<<<(N_ * N_) / 256, 256, 0, stream>>>(sn_bf, ad_bf, adj_bf,
                                                     gsum, cs, cb);
    // g = concat(xh, adj@xh, adj@adj@xh); hop outputs stored transposed
    k_pack_xh<<<nPackXh, 256, 0, stream>>>(xsrc, t, T, h_in, nullptr, xh_bf);
    k_gemm_bf16<<<gHop, 256, 0, stream>>>(adj_bf, N_, xh_bf, N_, hop_bf, N_,
                                          N_, nullptr, 3);
    k_gemm_bf16<<<gHop, 256, 0, stream>>>(adj_bf, N_, hop_bf, N_, hop2_f, N_,
                                          N_, nullptr, 4);
    k_pack_g<<<nPackG, 256, 0, stream>>>(xh_bf, hop_bf, hop2_f, g_bf);
    // z, r with fused bias+sigmoid epilogue
    k_gemm_bf16<<<gGate, 256, 0, stream>>>(g_bf, KP_, wz, KP_, zbuf, H_, KP_,
                                           bz, 1);
    k_gemm_bf16<<<gGate, 256, 0, stream>>>(g_bf, KP_, wr, KP_, rbuf, H_, KP_,
                                           br, 1);
    // candidate path with gated hidden state
    k_pack_xh<<<nPackXh, 256, 0, stream>>>(xsrc, t, T, h_in, rbuf, xh_bf);
    k_gemm_bf16<<<gHop, 256, 0, stream>>>(adj_bf, N_, xh_bf, N_, hop_bf, N_,
                                          N_, nullptr, 3);
    k_gemm_bf16<<<gHop, 256, 0, stream>>>(adj_bf, N_, hop_bf, N_, hop2_f, N_,
                                          N_, nullptr, 4);
    k_pack_g<<<nPackG, 256, 0, stream>>>(xh_bf, hop_bf, hop2_f, g_bf);
    k_gemm_bf16<<<gGate, 256, 0, stream>>>(g_bf, KP_, wh, KP_, htbuf, H_, KP_,
                                           bh, 2);
    // h = LN((1-z)h + z h~) * gamma + beta
    k_update_ln<<<M2_ / 8, rowsBlk, 0, stream>>>(zbuf, htbuf, h_in, gamma,
                                                 beta, h_out);
  };

  float* hc = h_a;
  float* hn = h_b;

  // encoder
  for (int t = 0; t < T_; ++t) {
    cell(X, t, T_, wp[0], enc_bz, wp[1], enc_br, wp[2], enc_bh, enc_g, enc_b,
         hc, hn);
    float* tmp = hc; hc = hn; hn = tmp;
  }

  // decoder
  hipMemsetAsync(ybuf, 0, (size_t)B_ * N_ * 4, stream);
  for (int s = 0; s < HOR_; ++s) {
    cell(ybuf, 0, 1, wp[3], dec_bz, wp[4], dec_br, wp[5], dec_bh, dec_g, dec_b,
         hc, hn);
    k_project<<<M2_ / 8, rowsBlk, 0, stream>>>(hn, proj_w, proj_b, out, ybuf,
                                               s);
    float* tmp = hc; hc = hn; hn = tmp;
  }
}
